// HEPT_23132693856576
// MI455X (gfx1250) — matrix operand order repaired and checked by reading, not compiled, not run on hardware
//
#include <hip/hip_runtime.h>
#include <hip/hip_fp16.h>

#define BS   128
#define DQK  67
#define DV   64
#define QKS  104   // Qs/Ks LDS row stride in f16 elems (208B = 52 dw, gcd(52,64)=4)
#define VTS  136   // Vt LDS row stride (272B = 68 dw)
#define PSR  136   // P  LDS row stride

// LDS offsets (bytes), all 16B aligned
#define OFF_QS   0
#define OFF_KS   26624     // 128*104*2
#define OFF_VT   53248     // +128*104*2
#define OFF_PM   70656     // +64*136*2
#define OFF_QSQ  105472    // +128*136*2
#define OFF_KSQ  105984
#define SMEM_BYTES 106496

typedef __attribute__((ext_vector_type(16))) _Float16 v16h;
typedef __attribute__((ext_vector_type(8)))  _Float16 v8h;
typedef __attribute__((ext_vector_type(8)))  float    v8f;

// A-fragment (16x32 f16, ISA 7.12.2): lane holds K = koff..koff+7 and koff+16..koff+23
__device__ __forceinline__ v16h ld_frag_a(const _Float16* base) {
  v8h lo = *(const v8h*)(base);
  v8h hi = *(const v8h*)(base + 16);
  v16h a;
#pragma unroll
  for (int i = 0; i < 8; ++i) { a[i] = lo[i]; a[i + 8] = hi[i]; }
  return a;
}

// B-fragment (32x16 f16): lane holds K = koff..koff+7 and koff+16..koff+23 of staged row n
__device__ __forceinline__ v16h ld_frag_b(const _Float16* base) {
  v8h lo = *(const v8h*)(base);
  v8h hi = *(const v8h*)(base + 16);
  v16h b;
#pragma unroll
  for (int i = 0; i < 8; ++i) { b[i] = lo[i]; b[i + 8] = hi[i]; }
  return b;
}

__global__ __launch_bounds__(256) void hept_rbf_attn(
    const float* __restrict__ Qg, const float* __restrict__ Kg,
    const float* __restrict__ Vg, float* __restrict__ Og)
{
  extern __shared__ unsigned char smem[];
  _Float16* Qs  = (_Float16*)(smem + OFF_QS);   // [128][QKS]
  _Float16* Ks  = (_Float16*)(smem + OFF_KS);   // [128][QKS]
  _Float16* Vt  = (_Float16*)(smem + OFF_VT);   // [64][VTS]  (V transposed)
  _Float16* Pm  = (_Float16*)(smem + OFF_PM);   // [128][PSR]
  float*    qsq = (float*)(smem + OFF_QSQ);     // [128]
  float*    ksq = (float*)(smem + OFF_KSQ);     // [128]

  const int tid  = threadIdx.x;
  const int lane = tid & 31;
  const int wv   = tid >> 5;      // 0..7 : wave owns S rows [16wv, 16wv+16)
  const int hs   = lane >> 4;     // half-wave select
  const int l16  = lane & 15;

  const size_t blk = blockIdx.x;
  const float* q = Qg + blk * (size_t)(BS * DQK);
  const float* k = Kg + blk * (size_t)(BS * DQK);
  const float* v = Vg + blk * (size_t)(BS * DV);
  float*       o = Og + blk * (size_t)(BS * DV);

  // ---- phase 0: zero padded staging + norm accumulators ----
  {
    unsigned int* z0 = (unsigned int*)Qs;
    unsigned int* z1 = (unsigned int*)Ks;
    const int ndw = BS * QKS / 2;   // 6656 dwords each
    for (int i = tid; i < ndw; i += 256) { z0[i] = 0u; z1[i] = 0u; }
    if (tid < BS) { qsq[tid] = 0.0f; ksq[tid] = 0.0f; }
  }
  __syncthreads();

  // ---- phase 1: stage Q,K (f32->f16) + f32 row sum-of-squares; stage V transposed ----
  for (int i = tid; i < BS * DQK; i += 256) {
    int r = i / DQK, c = i - r * DQK;
    float qv = q[i], kv = k[i];
    Qs[r * QKS + c] = (_Float16)qv;
    Ks[r * QKS + c] = (_Float16)kv;
    atomicAdd(&qsq[r], qv * qv);
    atomicAdd(&ksq[r], kv * kv);
  }
  for (int i = tid; i < BS * DV; i += 256) {
    int r = i >> 6, c = i & 63;
    Vt[c * VTS + r] = (_Float16)v[i];
  }
  __syncthreads();
  if (tid < BS) { qsq[tid] *= -0.5f; ksq[tid] *= -0.5f; }
  __syncthreads();

  // ---- phase 2: S = Q K^T, epilogue exp(min(S - .5|q|^2 - .5|k|^2, 0)) -> P (f16 in LDS) ----
  const int m = 16 * wv + l16;                 // A-matrix row for this lane
  const v16h aq0 = ld_frag_a(&Qs[m * QKS +  0 + 8 * hs]);
  const v16h aq1 = ld_frag_a(&Qs[m * QKS + 32 + 8 * hs]);
  const v16h aq2 = ld_frag_a(&Qs[m * QKS + 64 + 8 * hs]);

#pragma unroll
  for (int tc = 0; tc < 8; ++tc) {
    v8f c = {};
    v16h b0 = ld_frag_b(&Ks[(16 * tc + l16) * QKS +  0 + 8 * hs]);
    c = __builtin_amdgcn_wmma_f32_16x16x32_f16(false, aq0, false, b0, (short)0, c, false, false);
    v16h b1 = ld_frag_b(&Ks[(16 * tc + l16) * QKS + 32 + 8 * hs]);
    c = __builtin_amdgcn_wmma_f32_16x16x32_f16(false, aq1, false, b1, (short)0, c, false, false);
    v16h b2 = ld_frag_b(&Ks[(16 * tc + l16) * QKS + 64 + 8 * hs]);
    c = __builtin_amdgcn_wmma_f32_16x16x32_f16(false, aq2, false, b2, (short)0, c, false, false);

    const float kn = ksq[16 * tc + l16];
#pragma unroll
    for (int r = 0; r < 8; ++r) {
      const int row = 16 * wv + r + 8 * hs;    // C/D layout: VGPR r = row r (+8 for upper half-wave)
      float x = c[r] + qsq[row] + kn;
      x = fminf(x, 0.0f);
      x = __builtin_amdgcn_exp2f(x * 1.44269504088896340f);  // exp(x) via v_exp_f32
      Pm[row * PSR + 16 * tc + l16] = (_Float16)x;           // P in [0,1]: exact f16 fit
    }
  }
  // No barrier: wave wv consumes exactly the P rows it produced.

  // ---- phase 3: O = P V ----
  const v16h ap0 = ld_frag_a(&Pm[m * PSR +  0 + 8 * hs]);
  const v16h ap1 = ld_frag_a(&Pm[m * PSR + 32 + 8 * hs]);
  const v16h ap2 = ld_frag_a(&Pm[m * PSR + 64 + 8 * hs]);
  const v16h ap3 = ld_frag_a(&Pm[m * PSR + 96 + 8 * hs]);

#pragma unroll
  for (int oc = 0; oc < 4; ++oc) {
    v8f acc = {};
    v16h b0 = ld_frag_b(&Vt[(16 * oc + l16) * VTS +  0 + 8 * hs]);
    acc = __builtin_amdgcn_wmma_f32_16x16x32_f16(false, ap0, false, b0, (short)0, acc, false, false);
    v16h b1 = ld_frag_b(&Vt[(16 * oc + l16) * VTS + 32 + 8 * hs]);
    acc = __builtin_amdgcn_wmma_f32_16x16x32_f16(false, ap1, false, b1, (short)0, acc, false, false);
    v16h b2 = ld_frag_b(&Vt[(16 * oc + l16) * VTS + 64 + 8 * hs]);
    acc = __builtin_amdgcn_wmma_f32_16x16x32_f16(false, ap2, false, b2, (short)0, acc, false, false);
    v16h b3 = ld_frag_b(&Vt[(16 * oc + l16) * VTS + 96 + 8 * hs]);
    acc = __builtin_amdgcn_wmma_f32_16x16x32_f16(false, ap3, false, b3, (short)0, acc, false, false);

#pragma unroll
    for (int r = 0; r < 8; ++r) {
      const int row = 16 * wv + r + 8 * hs;
      o[row * DV + 16 * oc + l16] = acc[r];    // lanes 0..15 contiguous -> coalesced
    }
  }
}

extern "C" void kernel_launch(void* const* d_in, const int* in_sizes, int n_in,
                              void* d_out, int out_size, void* d_ws, size_t ws_size,
                              hipStream_t stream) {
  const float* q = (const float*)d_in[0];
  const float* k = (const float*)d_in[1];
  const float* v = (const float*)d_in[2];
  float* o = (float*)d_out;

  const int nblk = in_sizes[0] / (BS * DQK);   // B*H*NB = 2048

  static int attr_set = 0;  // host-side attribute config only (no stream work); idempotent
  if (!attr_set) {
    (void)hipFuncSetAttribute((const void*)hept_rbf_attn,
                              hipFuncAttributeMaxDynamicSharedMemorySize, SMEM_BYTES);
    attr_set = 1;
  }

  hept_rbf_attn<<<dim3(nblk), dim3(256), SMEM_BYTES, stream>>>(q, k, v, o);
}
